// BertWordPair_13202729468188
// MI455X (gfx1250) — matrix operand order repaired and checked by reading, not compiled, not run on hardware
//
#include <hip/hip_runtime.h>
#include <hip/hip_bf16.h>
#include <math.h>

typedef __attribute__((ext_vector_type(16))) __bf16 v16bf;
typedef __attribute__((ext_vector_type(8)))  __bf16 v8bf;
typedef __attribute__((ext_vector_type(8)))  float  v8f;
typedef __attribute__((ext_vector_type(4)))  unsigned int u32x4;
typedef __attribute__((ext_vector_type(8)))  int i32x8;
typedef __attribute__((ext_vector_type(4)))  int i32x4;

#define NEGV (-1000000000.0f)

__device__ inline v8f wmma_bf16(v16bf a, v16bf b, v8f c) {
  return __builtin_amdgcn_wmma_f32_16x16x32_bf16(false, a, false, b, (short)0, c,
                                                 false, false);
}

// ---------------------------------------------------------------------------
// TDM 2D tile load: global (rows x dim0 f32, row stride stride0 elements)
// -> LDS at byte offset ldsOff, packed contiguously. Per CDNA5 D# layout
// (CDNA5 ISA): group0 = {flags, lds_addr, global_addr,
// type=2}; group1 = {data_size=4B, tensor dims == tile dims, stride}.
// Groups 2/3/4 zero (2D tensor, no gather/iterate). Issued once per wave;
// tracked by TENSORcnt. 6-arg builtin form (clang-23 / therock headers).
// ---------------------------------------------------------------------------
__device__ inline void tdm_load_2d(unsigned ldsOff, const void* gptr,
                                   unsigned dim0, unsigned rows,
                                   unsigned stride0) {
  unsigned long long ga = (unsigned long long)(uintptr_t)gptr;
  u32x4 g0;
  g0[0] = 1u;                                              // count=1 (valid)
  g0[1] = ldsOff;                                          // lds_addr [63:32]
  g0[2] = (unsigned)(ga & 0xFFFFFFFFu);                    // global_addr lo
  g0[3] = (unsigned)((ga >> 32) & 0x01FFFFFFu) | (2u << 30); // addr hi | type=2
  i32x8 g1;
  g1[0] = (int)(2u << 16);                                 // data_size=2 (4B)
  g1[1] = (int)((dim0 & 0xFFFFu) << 16);                   // tensor_dim0 lo16
  g1[2] = (int)(((dim0 >> 16) & 0xFFFFu) | ((rows & 0xFFFFu) << 16)); // d0 hi | d1 lo
  g1[3] = (int)(((rows >> 16) & 0xFFFFu) | ((dim0 & 0xFFFFu) << 16)); // d1 hi | tile_dim0
  g1[4] = (int)(rows & 0xFFFFu);                           // tile_dim1 (tile_dim2=0)
  g1[5] = (int)stride0;                                    // dim0_stride lo32
  g1[6] = 0;
  g1[7] = 0;
  i32x4 z4 = {0, 0, 0, 0};
  i32x8 z8 = {0, 0, 0, 0, 0, 0, 0, 0};
  __builtin_amdgcn_tensor_load_to_lds(g0, g1, z4, z4, z8, 0);
}

// ---------------------------------------------------------------------------
// GEMM: C[M,N] = A[M,K] @ W[K,N] + bias[N], fp32 in / fp32 out, bf16 WMMA.
// Block = 128 threads (4 waves); macro tile 64x64; K-step 32.
// Double-buffered TDM pipeline: wave w DMA-loads A rows [16w,16w+16) and
// W k-rows [8w,8w+8) of the next tile while converting/computing the current.
// Requires 64|M, 64|N, 32|K (holds for all uses here).
// ---------------------------------------------------------------------------
__global__ __launch_bounds__(128)
void gemm_bias_bf16(const float* __restrict__ A, const float* __restrict__ W,
                    const float* __restrict__ bias, float* __restrict__ C,
                    int M, int N, int K) {
  __shared__ __align__(16) float  Asta[2][64 * 32];   // raw f32 A tiles (TDM dest)
  __shared__ __align__(16) float  Bsta[2][32 * 64];   // raw f32 W tiles (TDM dest)
  __shared__ __align__(32) __bf16 Abf[2][64 * 32];    // bf16 [row][k]
  __shared__ __align__(32) __bf16 Bbf[2][64 * 32];    // bf16 [n][k] (transposed)

  const int t = threadIdx.x;
  const int lane = t & 31;
  const int wave = t >> 5;
  const int lm = lane & 15;
  const int hi = lane >> 4;
  const int m0 = blockIdx.x * 64;
  const int n0 = blockIdx.y * 64;

  v8f acc[4] = {};

  auto issue = [&](int buf, int k0) {
    unsigned offA = (unsigned)(uintptr_t)(void*)&Asta[buf][wave * 16 * 32];
    tdm_load_2d(offA, A + (size_t)(m0 + wave * 16) * K + k0, 32u, 16u,
                (unsigned)K);
    unsigned offB = (unsigned)(uintptr_t)(void*)&Bsta[buf][wave * 8 * 64];
    tdm_load_2d(offB, W + (size_t)(k0 + wave * 8) * N + n0, 64u, 8u,
                (unsigned)N);
  };

  issue(0, 0);
  const int KS = K / 32;
  for (int s = 0; s < KS; ++s) {
    const int cur = s & 1;
    if (s + 1 < KS) {
      issue(cur ^ 1, (s + 1) * 32);
      __builtin_amdgcn_s_wait_tensorcnt(2);   // pair for step s complete
    } else {
      __builtin_amdgcn_s_wait_tensorcnt(0);
    }
    __syncthreads();                          // all waves' DMA visible

    // f32 stage -> bf16 tiles (B transposed to [n][k])
    #pragma unroll
    for (int i = 0; i < 16; ++i) {
      int e = t + i * 128;
      Abf[cur][e] = (__bf16)Asta[cur][e];
      int kk = e >> 6, n = e & 63;
      Bbf[cur][n * 32 + kk] = (__bf16)Bsta[cur][e];
    }
    __syncthreads();

    // WMMA fragments: A lane lm -> row, K pattern hi*16+{0..15}
    int ar = wave * 16 + lm;
    int kb = hi * 16;
    v16bf afrag = *(const v16bf*)&Abf[cur][ar * 32 + kb];
    #pragma unroll
    for (int j = 0; j < 4; ++j) {
      int n = j * 16 + lm;
      v16bf bfrag = *(const v16bf*)&Bbf[cur][n * 32 + hi * 16];
      acc[j] = wmma_bf16(afrag, bfrag, acc[j]);
    }
  }

  #pragma unroll
  for (int j = 0; j < 4; ++j) {
    int n = n0 + j * 16 + lm;
    float bv = bias ? bias[n] : 0.0f;
    #pragma unroll
    for (int r = 0; r < 8; ++r) {
      int m = m0 + wave * 16 + r + hi * 8;
      C[(size_t)m * N + n] = acc[j][r] + bv;
    }
  }
}

// ---------------------------------------------------------------------------
// Flash attention: one wave per (b, head, 16-query-row tile).
// q/k/v layout: [b*S + s][h*64 + d] (row stride 768). Mask: [b][m][n] int.
// Online softmax; P restaged via LDS to WMMA-A layout.
// ---------------------------------------------------------------------------
__global__ __launch_bounds__(32)
void flash_attn(const float* __restrict__ q, const float* __restrict__ k,
                const float* __restrict__ v, const int* __restrict__ mask,
                float* __restrict__ ctx, int S, int heads) {
  __shared__ __align__(16) float Plds[16 * 32];
  const int bid = blockIdx.x;
  const int mt = bid & 31;                  // S/16 == 32
  const int h = (bid >> 5) % heads;
  const int b = bid / (32 * heads);
  const int lane = threadIdx.x;
  const int lm = lane & 15;
  const int hi = lane >> 4;
  const int m0 = mt * 16;
  const int HC = 768, HD = 64;

  const float* qrow = q + ((size_t)b * S + m0 + lm) * HC + h * HD;
  v16bf qa[2];
  #pragma unroll
  for (int f = 0; f < 2; ++f) {
    int kb = f * 32 + hi * 16;
    float tmp[16];
    #pragma unroll
    for (int i = 0; i < 8; ++i) { tmp[i] = qrow[kb + i]; tmp[i + 8] = qrow[kb + 8 + i]; }
    v16bf tv;
    #pragma unroll
    for (int i = 0; i < 16; ++i) tv[i] = (__bf16)tmp[i];
    qa[f] = tv;
  }

  v8f acc[4] = {};
  float rowm[8], rowl[8];
  #pragma unroll
  for (int r = 0; r < 8; ++r) { rowm[r] = -1e30f; rowl[r] = 0.0f; }
  const int* mbase = mask + (size_t)b * S * S;

  for (int n0 = 0; n0 < S; n0 += 32) {
    v8f sc[2];
    #pragma unroll
    for (int ct = 0; ct < 2; ++ct) {
      const float* krow = k + ((size_t)b * S + n0 + ct * 16 + lm) * HC + h * HD;
      v8f cz = {};
      #pragma unroll
      for (int kh = 0; kh < 2; ++kh) {
        int kb = kh * 32 + hi * 16;
        float tmp[16];
        #pragma unroll
        for (int i = 0; i < 16; ++i) tmp[i] = krow[kb + i];
        v16bf bf;
        #pragma unroll
        for (int i = 0; i < 16; ++i) bf[i] = (__bf16)tmp[i];
        cz = wmma_bf16(qa[kh], bf, cz);
      }
      sc[ct] = cz;
    }
    // scale + mask
    #pragma unroll
    for (int ct = 0; ct < 2; ++ct) {
      int n = n0 + ct * 16 + lm;
      #pragma unroll
      for (int r = 0; r < 8; ++r) {
        int m = m0 + r + hi * 8;
        float s = sc[ct][r] * 0.125f;
        if (mbase[(size_t)m * S + n] <= 0) s = NEGV;
        sc[ct][r] = s;
      }
    }
    // online softmax per row (each row lives in one 16-lane half)
    #pragma unroll
    for (int r = 0; r < 8; ++r) {
      float mx = fmaxf(sc[0][r], sc[1][r]);
      #pragma unroll
      for (int o = 1; o < 16; o <<= 1) mx = fmaxf(mx, __shfl_xor(mx, o, 32));
      float nm = fmaxf(rowm[r], mx);
      float alpha = __expf(rowm[r] - nm);
      float p0 = __expf(sc[0][r] - nm);
      float p1 = __expf(sc[1][r] - nm);
      float ls = p0 + p1;
      #pragma unroll
      for (int o = 1; o < 16; o <<= 1) ls += __shfl_xor(ls, o, 32);
      rowl[r] = rowl[r] * alpha + ls;
      rowm[r] = nm;
      #pragma unroll
      for (int cc = 0; cc < 4; ++cc) acc[cc][r] *= alpha;
      Plds[(r + hi * 8) * 32 + lm] = p0;
      Plds[(r + hi * 8) * 32 + 16 + lm] = p1;
    }
    __syncthreads();
    // P as WMMA-A fragment (16x32)
    v16bf pf;
    {
      const float* pr = &Plds[lm * 32];
      int kb = hi * 16;
      #pragma unroll
      for (int i = 0; i < 8; ++i) {
        pf[i] = (__bf16)pr[kb + i];
        pf[i + 8] = (__bf16)pr[kb + 8 + i];
      }
    }
    // ctx += P @ V (4 output 16-col chunks); batch strided loads, then cvt
    #pragma unroll
    for (int cc = 0; cc < 4; ++cc) {
      float tmp[16];
      #pragma unroll
      for (int i = 0; i < 16; ++i) {
        int vr = n0 + hi * 16 + i;
        tmp[i] = v[((size_t)b * S + vr) * HC + h * HD + cc * 16 + lm];
      }
      v16bf vf;
      #pragma unroll
      for (int i = 0; i < 16; ++i) vf[i] = (__bf16)tmp[i];
      acc[cc] = wmma_bf16(pf, vf, acc[cc]);
    }
    __syncthreads();
  }

  #pragma unroll
  for (int cc = 0; cc < 4; ++cc) {
    #pragma unroll
    for (int r = 0; r < 8; ++r) {
      int m = m0 + r + hi * 8;
      ctx[((size_t)b * S + m) * HC + h * HD + cc * 16 + lm] = acc[cc][r] / rowl[r];
    }
  }
}

// ---------------------------------------------------------------------------
__global__ void max3_kernel(const float* __restrict__ a, const float* __restrict__ b,
                            const float* __restrict__ c, float* __restrict__ o, int n) {
  int i = blockIdx.x * blockDim.x + threadIdx.x;
  if (i < n) o[i] = fmaxf(fmaxf(a[i], b[i]), c[i]);
}

// ---------------------------------------------------------------------------
// RoPE prep: reads proj (B,S,C,256) fp32, writes roped bf16 in [b][c][s][64].
// part selects qt(0)/qu(64)/kt(128)/ku(192); posSign is +/-1.
// ---------------------------------------------------------------------------
__global__ __launch_bounds__(256)
void rope_prep(const float* __restrict__ proj, const int* __restrict__ pos,
               int part, int posSign, float base, int C,
               __bf16* __restrict__ out, int S, int n) {
  int idx = blockIdx.x * blockDim.x + threadIdx.x;
  if (idx >= n) return;
  int j = idx & 31;
  int sc = idx >> 5;
  int c = sc % C;
  int bs = sc / C;
  int s = bs % S;
  int b = bs / S;
  float p = (float)(posSign * pos[b * S + s]);
  float fr = __powf(base, -2.0f * (float)j / 64.0f);
  float ang = p * fr;
  float sn = __sinf(ang), cs = __cosf(ang);
  const float* src = proj + (((size_t)b * S + s) * C + c) * 256 + part + 2 * j;
  float v0 = src[0], v1 = src[1];
  __bf16* dst = out + (((size_t)b * C + c) * S + s) * 64 + 2 * j;
  dst[0] = (__bf16)(v0 * cs - v1 * sn);
  dst[1] = (__bf16)(v1 * cs + v0 * sn);
}

// ---------------------------------------------------------------------------
// Pairwise RoPE logits: one wave per (b, c, 16x16 tile of (m,n)).
// ---------------------------------------------------------------------------
__global__ __launch_bounds__(32)
void pair_logits(const __bf16* __restrict__ qtp, const __bf16* __restrict__ qtn,
                 const __bf16* __restrict__ ktp, const __bf16* __restrict__ ktn,
                 const __bf16* __restrict__ qup, const __bf16* __restrict__ qun,
                 const __bf16* __restrict__ kup, const __bf16* __restrict__ kun,
                 const int* __restrict__ tids, int C, int useUtt,
                 float* __restrict__ out, int S) {
  const int lane = threadIdx.x;
  const int lm = lane & 15;
  const int hi = lane >> 4;
  const int nt = S / 16;
  const int m0 = (blockIdx.x / nt) * 16;
  const int n0 = (blockIdx.x % nt) * 16;
  const int c = blockIdx.y;
  const int b = blockIdx.z;
  const size_t rowBase = ((size_t)b * C + c) * S;

  auto loadA = [&](const __bf16* Q, int f) -> v16bf {
    const __bf16* p = Q + (rowBase + m0 + lm) * 64 + f * 32 + hi * 16;
    return *(const v16bf*)p;
  };
  auto loadB = [&](const __bf16* Kk, int f) -> v16bf {
    const __bf16* p = Kk + (rowBase + n0 + lm) * 64 + f * 32 + hi * 16;
    return *(const v16bf*)p;
  };

  v8f Lpp = {}, Lnp = {}, Lpn = {};
  {
    v16bf qp0 = loadA(qtp, 0), qp1 = loadA(qtp, 1);
    v16bf qn0 = loadA(qtn, 0), qn1 = loadA(qtn, 1);
    v16bf kp0 = loadB(ktp, 0), kp1 = loadB(ktp, 1);
    v16bf kn0 = loadB(ktn, 0), kn1 = loadB(ktn, 1);
    Lpp = wmma_bf16(qp0, kp0, Lpp); Lpp = wmma_bf16(qp1, kp1, Lpp);
    Lnp = wmma_bf16(qn0, kp0, Lnp); Lnp = wmma_bf16(qn1, kp1, Lnp);
    Lpn = wmma_bf16(qp0, kn0, Lpn); Lpn = wmma_bf16(qp1, kn1, Lpn);
  }
  v8f Upp = {}, Unp = {}, Upn = {};
  if (useUtt) {
    v16bf qp0 = loadA(qup, 0), qp1 = loadA(qup, 1);
    v16bf qn0 = loadA(qun, 0), qn1 = loadA(qun, 1);
    v16bf kp0 = loadB(kup, 0), kp1 = loadB(kup, 1);
    v16bf kn0 = loadB(kun, 0), kn1 = loadB(kun, 1);
    Upp = wmma_bf16(qp0, kp0, Upp); Upp = wmma_bf16(qp1, kp1, Upp);
    Unp = wmma_bf16(qn0, kp0, Unp); Unp = wmma_bf16(qn1, kp1, Unp);
    Upn = wmma_bf16(qp0, kn0, Upn); Upn = wmma_bf16(qp1, kn1, Upn);
  }

  int n = n0 + lm;
  int tj = tids[(size_t)b * S + n];
  #pragma unroll
  for (int r = 0; r < 8; ++r) {
    int m = m0 + r + hi * 8;
    int ti = tids[(size_t)b * S + m];
    float vv;
    if (ti > 0 && ti < tj)      vv = Lnp[r];
    else if (tj > 0 && tj < ti) vv = Lpn[r];
    else                        vv = Lpp[r];
    if (useUtt) {
      float uu;
      if (ti > 0 && ti < tj)      uu = Unp[r];
      else if (tj > 0 && tj < ti) uu = Upn[r];
      else                        uu = Upp[r];
      vv += uu;
    }
    out[(((size_t)b * S + m) * S + n) * C + c] = vv;
  }
}

// ---------------------------------------------------------------------------
__global__ void zero2(float* acc) {
  if (blockIdx.x == 0 && threadIdx.x == 0) { acc[0] = 0.0f; acc[1] = 0.0f; }
}

__global__ __launch_bounds__(256)
void loss_reduce(const float* __restrict__ logits, const int* __restrict__ labels,
                 const int* __restrict__ mask, int C, float lw, int n,
                 float* __restrict__ acc) {
  __shared__ float s0[256];
  __shared__ float s1[256];
  int idx = blockIdx.x * 256 + threadIdx.x;
  float wn = 0.0f, ww = 0.0f;
  if (idx < n) {
    const float* lg = logits + (size_t)idx * C;
    int lab = labels[idx];
    float mx = lg[0];
    for (int i = 1; i < C; ++i) mx = fmaxf(mx, lg[i]);
    float se = 0.0f;
    for (int i = 0; i < C; ++i) se += __expf(lg[i] - mx);
    float lse = __logf(se) + mx;
    float nll = lse - lg[lab];
    float w = (lab == 0 ? 1.0f : lw) * (mask[idx] == 1 ? 1.0f : 0.0f);
    wn = w * nll;
    ww = w;
  }
  s0[threadIdx.x] = wn;
  s1[threadIdx.x] = ww;
  __syncthreads();
  for (int o = 128; o > 0; o >>= 1) {
    if (threadIdx.x < o) {
      s0[threadIdx.x] += s0[threadIdx.x + o];
      s1[threadIdx.x] += s1[threadIdx.x + o];
    }
    __syncthreads();
  }
  if (threadIdx.x == 0) {
    atomicAdd(&acc[0], s0[0]);
    atomicAdd(&acc[1], s1[0]);
  }
}

__global__ void finalize_loss(const float* acc, float* outv) {
  if (blockIdx.x == 0 && threadIdx.x == 0) *outv = acc[0] / acc[1];
}

// ---------------------------------------------------------------------------
extern "C" void kernel_launch(void* const* d_in, const int* in_sizes, int n_in,
                              void* d_out, int out_size, void* d_ws, size_t ws_size,
                              hipStream_t stream) {
  const float* x     = (const float*)d_in[0];
  const float* attnW = (const float*)d_in[1];
  const float* attnb = (const float*)d_in[2];
  const float* W_ent = (const float*)d_in[3];
  const float* b_ent = (const float*)d_in[4];
  const float* W_rel = (const float*)d_in[5];
  const float* b_rel = (const float*)d_in[6];
  const float* W_pol = (const float*)d_in[7];
  const float* b_pol = (const float*)d_in[8];
  const int* reply    = (const int*)d_in[9];
  const int* speaker  = (const int*)d_in[10];
  const int* threadm  = (const int*)d_in[11];
  const int* sentence = (const int*)d_in[12];
  const int* fullm    = (const int*)d_in[13];
  const int* tok_idx  = (const int*)d_in[14];
  const int* utt_idx  = (const int*)d_in[15];
  const int* tids     = (const int*)d_in[16];
  const int* ent_m    = (const int*)d_in[17];
  const int* rel_m    = (const int*)d_in[18];
  const int* pol_m    = (const int*)d_in[19];

  const int B = 8, S = 512, H = 768, HEADS = 12;
  const int M = B * S;
  const size_t MS = (size_t)M * H;

  float* w = (float*)d_ws;
  float* qb   = w;
  float* kb2  = w + MS;
  float* vb   = w + 2 * MS;
  float* ctxb = w + 3 * MS;
  float* r0   = w + 4 * MS;
  float* r1   = w + 5 * MS;
  float* r2   = w + 6 * MS;
  float* hb   = w + 7 * MS;
  float* proj = w + 8 * MS;
  __bf16* rp = (__bf16*)(w + 8 * MS + (size_t)M * 1024);
  const size_t RSZ = (size_t)B * S * 4 * 64;
  __bf16* qtp = rp;
  __bf16* qtn = rp + RSZ;
  __bf16* ktp = rp + 2 * RSZ;
  __bf16* ktn = rp + 3 * RSZ;
  __bf16* qup = rp + 4 * RSZ;
  __bf16* qun = rp + 5 * RSZ;
  __bf16* kup = rp + 6 * RSZ;
  __bf16* kun = rp + 7 * RSZ;
  float* acc = (float*)(rp + 8 * RSZ);

  float* out = (float*)d_out;
  float* t0 = out + 3;
  float* t1 = t0 + (size_t)B * S * S * 4;
  float* t2 = t1 + (size_t)B * S * S * 3;

  auto G = [&](const float* A, const float* W_, const float* b_, float* C_, int N) {
    dim3 grid(M / 64, N / 64);
    gemm_bias_bf16<<<grid, 128, 0, stream>>>(A, W_, b_, C_, M, N, H);
  };

  // ---- three MHA branches: reply, thread, speaker masks ----
  const int* masks[3] = {reply, threadm, speaker};
  float* routs[3] = {r0, r1, r2};
  const size_t HH = (size_t)H * H;
  for (int br = 0; br < 3; ++br) {
    G(x, attnW + (size_t)(br * 4 + 0) * HH, attnb + (br * 4 + 0) * H, qb, H);
    G(x, attnW + (size_t)(br * 4 + 1) * HH, attnb + (br * 4 + 1) * H, kb2, H);
    G(x, attnW + (size_t)(br * 4 + 2) * HH, attnb + (br * 4 + 2) * H, vb, H);
    flash_attn<<<B * HEADS * (S / 16), 32, 0, stream>>>(qb, kb2, vb, masks[br],
                                                        ctxb, S, HEADS);
    G(ctxb, attnW + (size_t)(br * 4 + 3) * HH, attnb + (br * 4 + 3) * H,
      routs[br], H);
  }
  {
    int n = M * H;
    max3_kernel<<<(n + 255) / 256, 256, 0, stream>>>(r0, r1, r2, hb, n);
  }

  // ---- classification heads ----
  auto classify = [&](const float* Wc, const float* bc, int C, const int* labm,
                      const int* mskm, int useUtt, float* logits_out,
                      float* loss_out) {
    G(hb, Wc, bc, proj, C * 256);
    int n = B * S * C * 32;
    int blocks = (n + 255) / 256;
    rope_prep<<<blocks, 256, 0, stream>>>(proj, tok_idx, 0,   +1, 10000.0f, C, qtp, S, n);
    rope_prep<<<blocks, 256, 0, stream>>>(proj, tok_idx, 0,   -1, 10000.0f, C, qtn, S, n);
    rope_prep<<<blocks, 256, 0, stream>>>(proj, tok_idx, 128, +1, 10000.0f, C, ktp, S, n);
    rope_prep<<<blocks, 256, 0, stream>>>(proj, tok_idx, 128, -1, 10000.0f, C, ktn, S, n);
    if (useUtt) {
      rope_prep<<<blocks, 256, 0, stream>>>(proj, utt_idx, 64,  +1, 15.0f, C, qup, S, n);
      rope_prep<<<blocks, 256, 0, stream>>>(proj, utt_idx, 64,  -1, 15.0f, C, qun, S, n);
      rope_prep<<<blocks, 256, 0, stream>>>(proj, utt_idx, 192, +1, 15.0f, C, kup, S, n);
      rope_prep<<<blocks, 256, 0, stream>>>(proj, utt_idx, 192, -1, 15.0f, C, kun, S, n);
    }
    dim3 pg((S / 16) * (S / 16), C, B);
    pair_logits<<<pg, 32, 0, stream>>>(qtp, qtn, ktp, ktn, qup, qun, kup, kun,
                                       tids, C, useUtt, logits_out, S);
    zero2<<<1, 32, 0, stream>>>(acc);
    int ne = B * S * S;
    loss_reduce<<<(ne + 255) / 256, 256, 0, stream>>>(logits_out, labm, mskm, C,
                                                      2.0f, ne, acc);
    finalize_loss<<<1, 32, 0, stream>>>(acc, loss_out);
  };

  classify(W_ent, b_ent, 4, ent_m, sentence, 0, t0, out + 0);
  classify(W_rel, b_rel, 3, rel_m, fullm,    1, t1, out + 1);
  classify(W_pol, b_pol, 4, pol_m, fullm,    1, t2, out + 2);
}
